// DiffPool_GNN_30648886624415
// MI455X (gfx1250) — matrix operand order repaired and checked by reading, not compiled, not run on hardware
//
#include <hip/hip_runtime.h>
#include <hip/hip_bf16.h>

// ---------------- problem constants (fixed by reference) ----------------
#define BN    16
#define MAXN  2048
#define INDIM 128
#define HIDD  64
#define KC1   205
#define KC1P  208      // K1 padded to multiple of 16
#define KC2   21
#define KC2P  32       // K2 padded
#define OUTD  2
#define OUTP  16       // OUT padded
#define NT1   17600    // total g1 nodes
#define NT2   14400    // total g2 nodes
#define NE1   281600
#define NE2   230400
#define NEH   32768

typedef _Float16 v16h __attribute__((ext_vector_type(16)));
typedef float    v8f  __attribute__((ext_vector_type(8)));

// ---------------- WMMA helpers (CDNA5 v_wmma_f32_16x16x32_f16) ----------------
__device__ inline v8f wmma16(v16h a, v16h b, v8f c) {
  return __builtin_amdgcn_wmma_f32_16x16x32_f16(
      /*neg_a=*/false, a, /*neg_b=*/false, b,
      /*c_mod=*/(short)0, c, /*reuse_a=*/false, /*reuse_b=*/false);
}

// A-matrix 16x32 f16 fragment (ISA 7.12.2): lane l holds row m=l&15;
// lanes 0-15: K pairs {0,1..6,7, 16..23}; lanes 16-31: {8..15, 24..31}.
// Non-transposed path loads K-pairs as float2 (k is always even -> 8B aligned).
__device__ inline v16h load_a_frag(const float* __restrict__ A, int lda,
                                   int m0, int kb, int trans, int lane) {
  int m = m0 + (lane & 15);
  int koff = (lane < 16) ? 0 : 8;
  v16h a;
  if (trans) {
#pragma unroll
    for (int v = 0; v < 8; ++v) {
      int k = kb + koff + ((v < 4) ? (2 * v) : (16 + 2 * (v - 4)));
      a[2 * v]     = (_Float16)A[(size_t)k * lda + m];
      a[2 * v + 1] = (_Float16)A[(size_t)(k + 1) * lda + m];
    }
  } else {
    const float* row = A + (size_t)m * lda;
#pragma unroll
    for (int v = 0; v < 8; ++v) {
      int k = kb + koff + ((v < 4) ? (2 * v) : (16 + 2 * (v - 4)));
      float2 f = *(const float2*)(row + k);           // global_load_b64
      a[2 * v]     = (_Float16)f.x;
      a[2 * v + 1] = (_Float16)f.y;
    }
  }
  return a;
}

// B-matrix 32x16 f16 fragment: lane l holds column n=l&15;
// lanes 0-15 carry K=kb+{0..7,16..23}, lanes 16-31 carry K=kb+{8..15,24..31}.
// Bounds handled branch-free: clamped index + multiplicative mask so every
// load is unconditional and in-bounds (no exec-mask churn).
__device__ inline v16h load_b_frag(const float* __restrict__ Bm, int ldb,
                                   int kb, int n0, int Kv, int Nv, int lane) {
  int n  = n0 + (lane & 15);
  int k0 = kb + ((lane < 16) ? 0 : 8);
  int   nc    = (n < Nv) ? n : (Nv - 1);
  float nmask = (n < Nv) ? 1.f : 0.f;
  v16h b;
#pragma unroll
  for (int i = 0; i < 16; ++i) {
    int k = k0 + i + ((i < 8) ? 0 : 8);
    int   kc = (k < Kv) ? k : (Kv - 1);
    float mk = (k < Kv) ? nmask : 0.f;
    float f  = Bm[(size_t)kc * ldb + nc] * mk;
    b[i] = (_Float16)f;
  }
  return b;
}

// ---------------- workspace layout (float/u32 words) ----------------
#define OFF_SD   ((size_t)0)                       // s_dense / s_softmax  [BN,MAXN,KC1P]
#define SZ_SD    ((size_t)BN * MAXN * KC1P)
#define OFF_XD   (OFF_SD + SZ_SD)                  // x_dense              [BN,MAXN,HIDD]
#define SZ_XD    ((size_t)BN * MAXN * HIDD)
#define OFF_AS   (OFF_XD + SZ_XD)                  // As = A @ s_soft      [BN,MAXN,KC1P]
#define OFF_BM   (OFF_AS + SZ_SD)                  // dedup bitmap         [BN,MAXN,64] u32
#define SZ_BM    ((size_t)BN * MAXN * 64)
#define OFF_A1   (OFF_BM + SZ_BM)                  // a1                   [BN,KC1P,KC1P]
#define SZ_A1    ((size_t)BN * KC1P * KC1P)
#define SZ_RH    ((size_t)BN * KC1P * HIDD)
#define OFF_X1   (OFF_A1 + SZ_A1)
#define OFF_X1E  (OFF_X1 + SZ_RH)
#define OFF_HE   (OFF_X1E + SZ_RH)
#define OFF_T64  (OFF_HE + SZ_RH)
#define SZ_RS    ((size_t)BN * KC1P * KC2P)
#define OFF_HS   (OFF_T64 + SZ_RH)
#define OFF_S2   (OFF_HS + SZ_RS)
#define OFF_U    (OFF_S2 + SZ_RS)
#define OFF_T32  (OFF_U + SZ_RS)
#define SZ_XH    ((size_t)BN * KC2P * HIDD)
#define OFF_X2   (OFF_T32 + SZ_RS)
#define OFF_HC   (OFF_X2 + SZ_XH)
#define SZ_A2M   ((size_t)BN * KC2P * KC2P)
#define OFF_A2   (OFF_HC + SZ_XH)
#define SZ_T16   ((size_t)BN * KC2P * OUTP)
#define OFF_T16  (OFF_A2 + SZ_A2M)
#define OFF_OUTN (OFF_T16 + SZ_T16)
#define WS_TOTAL (OFF_OUTN + SZ_T16)

// ---------------- kernels ----------------
__global__ void zero_ws_kernel(unsigned* __restrict__ p, size_t n) {
  size_t i = (size_t)blockIdx.x * blockDim.x + threadIdx.x;
  size_t st = (size_t)gridDim.x * blockDim.x;
  for (; i < n; i += st) p[i] = 0u;
}

// Level-1 projections: s = x @ W_pool, h = relu(x @ W_emb), scattered to dense slots.
// grid = (node_tiles, 17 col tiles), block = 32 (one wave, one 16x16 tile).
__global__ void proj_kernel(const float* __restrict__ x,
                            const float* __restrict__ Wpool,
                            const float* __restrict__ Wemb,
                            const int* __restrict__ batch,
                            const int* __restrict__ slc1,
                            const int* __restrict__ slc2,
                            int isG2,
                            float* __restrict__ sdense,
                            float* __restrict__ xdense) {
  int lane = threadIdx.x;
  int m0 = blockIdx.x * 16;
  int ct = blockIdx.y;
  int pool = (ct < 13) ? 1 : 0;
  const float* Bm = pool ? Wpool : Wemb;
  int ldb = pool ? KC1 : HIDD;
  int n0  = pool ? ct * 16 : (ct - 13) * 16;
  v8f acc = {};
#pragma unroll
  for (int kb = 0; kb < INDIM; kb += 32) {
    v16h a = load_a_frag(x, INDIM, m0, kb, 0, lane);
    v16h b = load_b_frag(Bm, ldb, kb, n0, INDIM, ldb, lane);
    acc = wmma16(a, b, acc);
  }
  int n  = n0 + (lane & 15);
  int mb = (lane < 16) ? 0 : 8;
#pragma unroll
  for (int v = 0; v < 8; ++v) {
    int node = m0 + mb + v;
    int bb = batch[node];
    int pos = isG2 ? (node - slc2[bb] + (slc1[bb + 1] - slc1[bb]))
                   : (node - slc1[bb]);
    size_t row = (size_t)bb * MAXN + pos;
    if (pool) {
      if (n < KC1) sdense[row * KC1P + n] = acc[v];
    } else {
      xdense[row * HIDD + n] = fmaxf(acc[v], 0.f);
    }
  }
}

// Row softmax over K1=205 cols, in place; zero the 3 pad cols. One wave per row.
__global__ void softmax_l1(float* __restrict__ s) {
  int wid  = (blockIdx.x * blockDim.x + threadIdx.x) >> 5;
  int lane = threadIdx.x & 31;
  if (wid >= BN * MAXN) return;
  float* p = s + (size_t)wid * KC1P;
  float mx = -3.4e38f;
  float ev[7];
#pragma unroll
  for (int i = 0; i < 7; ++i) { int k = lane + 32 * i; if (k < KC1) mx = fmaxf(mx, p[k]); }
#pragma unroll
  for (int o = 16; o > 0; o >>= 1) mx = fmaxf(mx, __shfl_xor(mx, o, 32));
  float sum = 0.f;
#pragma unroll
  for (int i = 0; i < 7; ++i) {
    int k = lane + 32 * i;
    float e = 0.f;
    if (k < KC1) e = __expf(p[k] - mx);
    ev[i] = e; sum += e;
  }
#pragma unroll
  for (int o = 16; o > 0; o >>= 1) sum += __shfl_xor(sum, o, 32);
  float inv = 1.f / sum;
#pragma unroll
  for (int i = 0; i < 7; ++i) { int k = lane + 32 * i; if (k < KC1) p[k] = ev[i] * inv; }
  if (lane < KC1P - KC1) p[KC1 + lane] = 0.f;
}

// Row softmax over K2=21 cols of s2 (rows < 205 only), zero pad cols.
__global__ void softmax_l2(float* __restrict__ s2) {
  int wid  = (blockIdx.x * blockDim.x + threadIdx.x) >> 5;
  int lane = threadIdx.x & 31;
  if (wid >= BN * KC1) return;
  int bb = wid / KC1, r = wid % KC1;
  float* p = s2 + ((size_t)bb * KC1P + r) * KC2P;
  float v = (lane < KC2) ? p[lane] : -3.4e38f;
  float mx = v;
  for (int o = 16; o > 0; o >>= 1) mx = fmaxf(mx, __shfl_xor(mx, o, 32));
  float e = (lane < KC2) ? __expf(v - mx) : 0.f;
  float sum = e;
  for (int o = 16; o > 0; o >>= 1) sum += __shfl_xor(sum, o, 32);
  p[lane] = (lane < KC2) ? (e / sum) : 0.f;
}

// Edge SpMM with bitmap dedup (reproduces adj.at[...].set(1.0) semantics):
// one wave per edge; lane 0 atomically test-and-sets the (src,dst) bit;
// on first insertion the wave scatter-adds s_soft[dst,:] into As[src,:].
__global__ void edge_spmm(const int* __restrict__ eg1, const int* __restrict__ eg2,
                          const int* __restrict__ ehh,
                          const int* __restrict__ b1, const int* __restrict__ b2,
                          const int* __restrict__ slc1, const int* __restrict__ slc2,
                          const float* __restrict__ ssoft,
                          float* __restrict__ As, unsigned* __restrict__ bitmap) {
  int w    = (blockIdx.x * blockDim.x + threadIdx.x) >> 5;
  int lane = threadIdx.x & 31;
  if (w >= NE1 + NE2 + NEH) return;
  int src, dst, bb, sp, dp;
  if (w < NE1) {
    src = eg1[w]; dst = eg1[NE1 + w];
    bb = b1[src]; int o = slc1[bb];
    sp = src - o; dp = dst - o;
  } else if (w < NE1 + NE2) {
    int e = w - NE1;
    src = eg2[e]; dst = eg2[NE2 + e];
    bb = b2[src]; int o = slc2[bb]; int n1 = slc1[bb + 1] - slc1[bb];
    sp = src - o + n1; dp = dst - o + n1;
  } else {
    int e = w - NE1 - NE2;
    src = ehh[e]; dst = ehh[NEH + e];
    bb = b1[src]; sp = src - slc1[bb];
    int n1 = slc1[bb + 1] - slc1[bb];
    dp = dst - slc2[bb] + n1;
  }
  int unique = 0;
  if (lane == 0) {
    unsigned* wp = bitmap + ((size_t)bb * MAXN + sp) * 64 + (dp >> 5);
    unsigned m = 1u << (dp & 31);
    unsigned old = atomicOr(wp, m);
    unique = ((old & m) == 0u) ? 1 : 0;
  }
  unique = __shfl(unique, 0, 32);
  if (!unique) return;
  const float* srow = ssoft + ((size_t)bb * MAXN + dp) * KC1P;
  float* arow = As + ((size_t)bb * MAXN + sp) * KC1P;
  for (int k = lane; k < KC1; k += 32) atomicAdd(&arow[k], srow[k]);
}

// Generic dual-term WMMA GEMM: D = act(A*B + [A2*B2]), batched over grid.z.
// One wave per 16x16 output tile; transA reads A as (K x M).
__global__ void gemm16(float* __restrict__ D, int ldd, long sD, int relu,
                       const float* __restrict__ A, int lda, long sA, int tA, int K,
                       const float* __restrict__ Bm, int ldb, long sB, int Kv, int Nv,
                       const float* __restrict__ A2, int lda2, long sA2, int tA2, int K2c,
                       const float* __restrict__ B2, int ldb2, long sB2, int Kv2, int Nv2) {
  int lane = threadIdx.x;
  int m0 = blockIdx.x * 16, n0 = blockIdx.y * 16, bz = blockIdx.z;
  const float* Ab = A + (size_t)bz * sA;
  const float* Bb = Bm + (size_t)bz * sB;
  v8f acc = {};
  for (int kb = 0; kb < K; kb += 32) {
    __builtin_prefetch(Bb + (size_t)(kb + 32) * ldb + n0, 0, 1);  // global_prefetch_b8
    v16h a = load_a_frag(Ab, lda, m0, kb, tA, lane);
    v16h b = load_b_frag(Bb, ldb, kb, n0, Kv, Nv, lane);
    acc = wmma16(a, b, acc);
  }
  if (A2 != nullptr) {
    const float* A2b = A2 + (size_t)bz * sA2;
    const float* B2b = B2 + (size_t)bz * sB2;
    for (int kb = 0; kb < K2c; kb += 32) {
      v16h a = load_a_frag(A2b, lda2, m0, kb, tA2, lane);
      v16h b = load_b_frag(B2b, ldb2, kb, n0, Kv2, Nv2, lane);
      acc = wmma16(a, b, acc);
    }
  }
  float* Db = D + (size_t)bz * sD;
  int n  = n0 + (lane & 15);
  int mb = (lane < 16) ? 0 : 8;
#pragma unroll
  for (int v = 0; v < 8; ++v) {
    float val = acc[v];
    if (relu) val = fmaxf(val, 0.f);
    Db[(size_t)(m0 + mb + v) * ldd + n] = val;
  }
}

// Mean over the 21 valid cluster rows -> [BN,2] f32 output.
__global__ void readout_kernel(const float* __restrict__ outn, float* __restrict__ out) {
  int b = blockIdx.x, lane = threadIdx.x;
  float v0 = (lane < KC2) ? outn[((size_t)b * KC2P + lane) * OUTP + 0] : 0.f;
  float v1 = (lane < KC2) ? outn[((size_t)b * KC2P + lane) * OUTP + 1] : 0.f;
  for (int o = 16; o > 0; o >>= 1) { v0 += __shfl_xor(v0, o, 32); v1 += __shfl_xor(v1, o, 32); }
  if (lane == 0) { out[b * 2 + 0] = v0 / (float)KC2; out[b * 2 + 1] = v1 / (float)KC2; }
}

// ---------------- host side ----------------
static inline void launch_gemm(hipStream_t st, float* D, int ldd, long sD, int relu,
                               int M, int N,
                               const float* A, int lda, long sA, int tA, int K,
                               const float* Bm, int ldb, long sB, int Kv, int Nv,
                               const float* A2 = nullptr, int lda2 = 0, long sA2 = 0,
                               int tA2 = 0, int K2c = 0,
                               const float* B2 = nullptr, int ldb2 = 0, long sB2 = 0,
                               int Kv2 = 0, int Nv2 = 0) {
  dim3 g(M / 16, N / 16, BN);
  gemm16<<<g, 32, 0, st>>>(D, ldd, sD, relu, A, lda, sA, tA, K, Bm, ldb, sB, Kv, Nv,
                           A2, lda2, sA2, tA2, K2c, B2, ldb2, sB2, Kv2, Nv2);
}

extern "C" void kernel_launch(void* const* d_in, const int* in_sizes, int n_in,
                              void* d_out, int out_size, void* d_ws, size_t ws_size,
                              hipStream_t stream) {
  const float* x_g1 = (const float*)d_in[0];
  const float* x_g2 = (const float*)d_in[1];
  const float* Wpool1 = (const float*)d_in[2];
  const float* Wpool2 = (const float*)d_in[3];
  const float* Wemb1 = (const float*)d_in[4];
  const float* Wemb2 = (const float*)d_in[5];
  const float* Wp1 = (const float*)d_in[6];
  const float* Up1 = (const float*)d_in[7];
  const float* Wp2 = (const float*)d_in[8];
  const float* Up2 = (const float*)d_in[9];
  const float* We1 = (const float*)d_in[10];
  const float* Ue1 = (const float*)d_in[11];
  const float* We2 = (const float*)d_in[12];
  const float* Ue2 = (const float*)d_in[13];
  const float* Wc1 = (const float*)d_in[14];
  const float* Uc1 = (const float*)d_in[15];
  const float* Wc2 = (const float*)d_in[16];
  const float* Uc2 = (const float*)d_in[17];
  const int* eg1 = (const int*)d_in[18];
  const int* eg2 = (const int*)d_in[19];
  const int* ehh = (const int*)d_in[20];
  const int* b1 = (const int*)d_in[21];
  const int* b2 = (const int*)d_in[22];
  const int* sl1 = (const int*)d_in[23];
  const int* sl2 = (const int*)d_in[24];

  float* ws = (float*)d_ws;
  float* SD = ws + OFF_SD;   float* XD = ws + OFF_XD;
  float* AS = ws + OFF_AS;   unsigned* BM = (unsigned*)(ws + OFF_BM);
  float* A1 = ws + OFF_A1;   float* X1 = ws + OFF_X1;
  float* X1E = ws + OFF_X1E; float* HE = ws + OFF_HE;
  float* T64 = ws + OFF_T64; float* HS = ws + OFF_HS;
  float* S2 = ws + OFF_S2;   float* UU = ws + OFF_U;
  float* T32 = ws + OFF_T32; float* X2 = ws + OFF_X2;
  float* HC = ws + OFF_HC;   float* A2M = ws + OFF_A2;
  float* T16 = ws + OFF_T16; float* OUTN = ws + OFF_OUTN;

  // 0) zero all workspace (adjacency bitmap + accumulators must be fresh)
  zero_ws_kernel<<<2048, 256, 0, stream>>>((unsigned*)d_ws, WS_TOTAL);

  // 1) level-1 projections (WMMA) + scatter to dense slots
  proj_kernel<<<dim3(NT1 / 16, 17), 32, 0, stream>>>(x_g1, Wpool1, Wemb1, b1, sl1, sl2, 0, SD, XD);
  proj_kernel<<<dim3(NT2 / 16, 17), 32, 0, stream>>>(x_g2, Wpool2, Wemb2, b2, sl1, sl2, 1, SD, XD);

  // 2) softmax over K1 (in place)
  softmax_l1<<<(BN * MAXN) / 8, 256, 0, stream>>>(SD);

  // 3) As = A @ s_soft via dedup'd edge scatter-add
  {
    long warps = (long)NE1 + NE2 + NEH;
    long blocks = (warps * 32 + 255) / 256;
    edge_spmm<<<(unsigned)blocks, 256, 0, stream>>>(eg1, eg2, ehh, b1, b2, sl1, sl2, SD, AS, BM);
  }

  long sSD = (long)MAXN * KC1P, sXD = (long)MAXN * HIDD;
  long sA1 = (long)KC1P * KC1P, sRH = (long)KC1P * HIDD, sRS = (long)KC1P * KC2P;
  long sXH = (long)KC2P * HIDD, sA2 = (long)KC2P * KC2P, sT16 = (long)KC2P * OUTP;

  // 4) a1 = s^T (A s) ; x1 = s^T x      (K = 2048 WMMA reductions)
  launch_gemm(stream, A1, KC1P, sA1, 0, KC1P, KC1P, SD, KC1P, sSD, 1, MAXN, AS, KC1P, sSD, MAXN, KC1P);
  launch_gemm(stream, X1, HIDD, sRH, 0, KC1P, HIDD, SD, KC1P, sSD, 1, MAXN, XD, HIDD, sXD, MAXN, HIDD);

  // 5) s2 = dense_gnn(x1, a1, Wp1, Up1, Wp2, Up2)
  launch_gemm(stream, T32, KC2P, sRS, 0, KC1P, KC2P, X1, HIDD, sRH, 0, HIDD, Wp1, KC2, 0, HIDD, KC2);
  launch_gemm(stream, HS, KC2P, sRS, 1, KC1P, KC2P, A1, KC1P, sA1, 0, KC1P, T32, KC2P, sRS, KC1P, KC2P,
              X1, HIDD, sRH, 0, HIDD, Up1, KC2, 0, HIDD, KC2);
  launch_gemm(stream, T32, KC2P, sRS, 0, KC1P, KC2P, HS, KC2P, sRS, 0, KC2P, Wp2, KC2, 0, KC2, KC2);
  launch_gemm(stream, S2, KC2P, sRS, 0, KC1P, KC2P, A1, KC1P, sA1, 0, KC1P, T32, KC2P, sRS, KC1P, KC2P,
              HS, KC2P, sRS, 0, KC2P, Up2, KC2, 0, KC2, KC2);

  // 6) x1e = dense_gnn(x1, a1, We1, Ue1, We2, Ue2)
  launch_gemm(stream, T64, HIDD, sRH, 0, KC1P, HIDD, X1, HIDD, sRH, 0, HIDD, We1, HIDD, 0, HIDD, HIDD);
  launch_gemm(stream, HE, HIDD, sRH, 1, KC1P, HIDD, A1, KC1P, sA1, 0, KC1P, T64, HIDD, sRH, KC1P, HIDD,
              X1, HIDD, sRH, 0, HIDD, Ue1, HIDD, 0, HIDD, HIDD);
  launch_gemm(stream, T64, HIDD, sRH, 0, KC1P, HIDD, HE, HIDD, sRH, 0, HIDD, We2, HIDD, 0, HIDD, HIDD);
  launch_gemm(stream, X1E, HIDD, sRH, 0, KC1P, HIDD, A1, KC1P, sA1, 0, KC1P, T64, HIDD, sRH, KC1P, HIDD,
              HE, HIDD, sRH, 0, HIDD, Ue2, HIDD, 0, HIDD, HIDD);

  // 7) softmax(s2); x2 = s2^T x1e; a2 = s2^T a1 s2
  softmax_l2<<<(BN * KC1 + 7) / 8, 256, 0, stream>>>(S2);
  launch_gemm(stream, X2, HIDD, sXH, 0, KC2P, HIDD, S2, KC2P, sRS, 1, KC1P, X1E, HIDD, sRH, KC1P, HIDD);
  launch_gemm(stream, UU, KC2P, sRS, 0, KC1P, KC2P, A1, KC1P, sA1, 0, KC1P, S2, KC2P, sRS, KC1P, KC2P);
  launch_gemm(stream, A2M, KC2P, sA2, 0, KC2P, KC2P, S2, KC2P, sRS, 1, KC1P, UU, KC2P, sRS, KC1P, KC2P);

  // 8) out_nodes = dense_gnn(x2, a2, Wc1, Uc1, Wc2, Uc2)
  launch_gemm(stream, T64, HIDD, sRH, 0, KC2P, HIDD, X2, HIDD, sXH, 0, HIDD, Wc1, HIDD, 0, HIDD, HIDD);
  launch_gemm(stream, HC, HIDD, sXH, 1, KC2P, HIDD, A2M, KC2P, sA2, 0, KC2P, T64, HIDD, sRH, KC2P, HIDD,
              X2, HIDD, sXH, 0, HIDD, Uc1, HIDD, 0, HIDD, HIDD);
  launch_gemm(stream, T16, OUTP, sT16, 0, KC2P, OUTP, HC, HIDD, sXH, 0, HIDD, Wc2, OUTD, 0, HIDD, OUTD);
  launch_gemm(stream, OUTN, OUTP, sT16, 0, KC2P, OUTP, A2M, KC2P, sA2, 0, KC2P, T16, OUTP, sT16, KC2P, OUTP,
              HC, HIDD, sXH, 0, HIDD, Uc2, OUTD, 0, HIDD, OUTD);

  // 9) mean over 21 clusters -> [16,2]
  readout_kernel<<<BN, 32, 0, stream>>>(OUTN, (float*)d_out);

  (void)in_sizes; (void)n_in; (void)out_size; (void)ws_size;
}
